// BiLSTMDecoder_19078244729269
// MI455X (gfx1250) — matrix operand order repaired and checked by reading, not compiled, not run on hardware
//
#include <hip/hip_runtime.h>
#include <hip/hip_bf16.h>
#include <math.h>

typedef __attribute__((ext_vector_type(16))) _Float16 v16h;
typedef __attribute__((ext_vector_type(8)))  _Float16 v8h;
typedef __attribute__((ext_vector_type(8)))  float    v8f;
typedef __attribute__((ext_vector_type(4)))  int      v4i;

#define BB 64
#define TT 512
#define DD 512
#define HH 2
#define HDD 256
#define KK 32

#if defined(__gfx1250__) && __has_builtin(__builtin_amdgcn_global_load_async_to_lds_b128)
#define HAVE_ASYNC_LDS 1
#else
#define HAVE_ASYNC_LDS 0
#endif

union V16H { v16h v; v8h h[2]; };

// ---------------- WMMA fragment loaders (16x16x32 f16) ----------------
// A fragment: 16(M) x 32(K) row-major tile. Lane l: m = l&15, kb = (l<16?0:8).
// elems 0..7  -> K = kb..kb+7 ; elems 8..15 -> K = kb+16..kb+23.
__device__ inline v16h load_a_frag(const _Float16* __restrict__ A, int lda, int m0, int k0) {
  int l  = threadIdx.x & 31;
  int m  = l & 15;
  int kb = (l < 16) ? 0 : 8;
  const _Float16* p = A + (size_t)(m0 + m) * lda + k0 + kb;
  V16H r;
  r.h[0] = *(const v8h*)(p);
  r.h[1] = *(const v8h*)(p + 16);
  return r.v;
}

// B fragment for C = A * W^T with W row-major [N, Kd]: B(k,n) = W[n][k].
// Lane l: n = l&15, kb = (l<16?0:8), elems 0..7 -> k = kb..kb+7 ; elems 8..15 -> k = kb+16..kb+23.
__device__ inline v16h load_b_frag(const _Float16* __restrict__ W, int ldw, int n0, int k0) {
  int l  = threadIdx.x & 31;
  int n  = l & 15;
  int kb = (l < 16) ? 0 : 8;
  const _Float16* p = W + (size_t)(n0 + n) * ldw + k0 + kb;
  V16H r;
  r.h[0] = *(const v8h*)(p);
  r.h[1] = *(const v8h*)(p + 16);
  return r.v;
}

// Software-pipelined (double-buffered) GEMM inner loop.
template<int NT>
__device__ inline void gemm_acc(const _Float16* __restrict__ A, int lda, int m0,
                                const _Float16* __restrict__ W, int ldw, int n0,
                                int Kd, v8f acc[NT]) {
  v16h a = load_a_frag(A, lda, m0, 0);
  v16h b[NT];
#pragma unroll
  for (int j = 0; j < NT; ++j) b[j] = load_b_frag(W, ldw, n0 + 16 * j, 0);
  for (int k0 = 32; k0 < Kd; k0 += 32) {
    v16h an = load_a_frag(A, lda, m0, k0);
    v16h bn[NT];
#pragma unroll
    for (int j = 0; j < NT; ++j) bn[j] = load_b_frag(W, ldw, n0 + 16 * j, k0);
#pragma unroll
    for (int j = 0; j < NT; ++j)
      acc[j] = __builtin_amdgcn_wmma_f32_16x16x32_f16(false, a, false, b[j],
                                                      (short)0, acc[j], false, false);
    a = an;
#pragma unroll
    for (int j = 0; j < NT; ++j) b[j] = bn[j];
  }
#pragma unroll
  for (int j = 0; j < NT; ++j)
    acc[j] = __builtin_amdgcn_wmma_f32_16x16x32_f16(false, a, false, b[j],
                                                    (short)0, acc[j], false, false);
}

// Cooperative stage of a 16x512 f16 A-tile (16KB) into LDS using the CDNA5
// async global->LDS DMA path (ASYNCcnt) when available. 128 threads/block.
__device__ inline void stage_a_tile(const _Float16* __restrict__ g, _Float16* s) {
#if HAVE_ASYNC_LDS
#pragma unroll
  for (int i = 0; i < 8; ++i) {
    int c = threadIdx.x + (i << 7);   // 1024 chunks of 16B
    __builtin_amdgcn_global_load_async_to_lds_b128(
        (__attribute__((address_space(1))) v4i*)(g + c * 8),
        (__attribute__((address_space(3))) v4i*)(s + c * 8), 0, 0);
  }
#if __has_builtin(__builtin_amdgcn_s_wait_asynccnt)
  __builtin_amdgcn_s_wait_asynccnt(0);
#else
  asm volatile("s_wait_asynccnt 0" ::: "memory");
#endif
#else
#pragma unroll
  for (int i = 0; i < 8; ++i) {
    int c = threadIdx.x + (i << 7);
    *(v8h*)(s + c * 8) = *(const v8h*)(g + c * 8);
  }
#endif
  __syncthreads();
}

// ---------------- f32 -> f16 conversion ----------------
__global__ void cvt_f32_f16_kernel(const float* __restrict__ in, _Float16* __restrict__ out, int n) {
  int i = blockIdx.x * blockDim.x + threadIdx.x;
  if (i < n) out[i] = (_Float16)in[i];
}

// ---------------- QKV projection GEMM + scatter ----------------
// One block per 16-row m-tile; A tile staged once in LDS via async DMA;
// each of 4 waves covers 6 groups of 64 output columns.
__global__ void __launch_bounds__(128, 1)
qkv_gemm_kernel(const _Float16* __restrict__ x16,
                const _Float16* __restrict__ w16,
                const float*    __restrict__ bias,
                _Float16* __restrict__ q16,
                _Float16* __restrict__ k16,
                _Float16* __restrict__ vt16) {
  __shared__ __align__(32) _Float16 atile[16 * DD];
  int m0 = blockIdx.x * 16;
  stage_a_tile(x16 + (size_t)m0 * DD, atile);

  int w = threadIdx.x >> 5;
  int l = threadIdx.x & 31;
  int col0  = l & 15;
  int rbase = (l < 16) ? 0 : 8;

  for (int g = 0; g < 6; ++g) {
    int n0 = (w * 6 + g) * 64;
    v8f acc[4];
#pragma unroll
    for (int j = 0; j < 4; ++j)
#pragma unroll
      for (int e = 0; e < 8; ++e) acc[j][e] = 0.f;

    gemm_acc<4>(atile, DD, 0, w16, DD, n0, DD, acc);  // A frags via ds_load

#pragma unroll
    for (int j = 0; j < 4; ++j) {
      int col  = n0 + j * 16 + col0;
      float bv = bias[col];
#pragma unroll
      for (int r = 0; r < 8; ++r) {
        int row = m0 + rbase + r;
        int b = row >> 9;          // row = b*T + t
        int t = row & (TT - 1);
        _Float16 hv = (_Float16)(acc[j][r] + bv);
        if (col < DD) {            // Q: [B,H,T,HD]
          int h = col >> 8, hd = col & (HDD - 1);
          q16[(((size_t)b * HH + h) * TT + t) * HDD + hd] = hv;
        } else if (col < 2 * DD) { // K: [B,H,T,HD]
          int c = col - DD; int h = c >> 8, hd = c & (HDD - 1);
          k16[(((size_t)b * HH + h) * TT + t) * HDD + hd] = hv;
        } else {                   // V transposed: [B,H,HD,T]
          int c = col - 2 * DD; int h = c >> 8, hd = c & (HDD - 1);
          vt16[(((size_t)b * HH + h) * HDD + hd) * TT + t] = hv;
        }
      }
    }
  }
}

// ---------------- Flash attention: one wave per (b,h,16 q-rows) ----------------
__global__ void __launch_bounds__(128, 1)
attn_kernel(const _Float16* __restrict__ q16,
            const _Float16* __restrict__ k16,
            const _Float16* __restrict__ vt16,
            _Float16* __restrict__ ctx16) {
  __shared__ __align__(32) _Float16 pst[4][16][32];   // per-wave P staging (C->A relayout)
  int wib  = threadIdx.x >> 5;
  int wave = (blockIdx.x * blockDim.x + threadIdx.x) >> 5;
  int qt = wave & 31;          // 32 q-tiles of 16 rows
  int h  = (wave >> 5) & 1;
  int b  = wave >> 6;
  int l  = threadIdx.x & 31;

  const _Float16* Q  = q16  + ((size_t)(b * HH + h) * TT + qt * 16) * HDD;
  const _Float16* Kp = k16  + (size_t)(b * HH + h) * TT * HDD;
  const _Float16* Vt = vt16 + (size_t)(b * HH + h) * HDD * TT;

  v8f acc[16];                 // 16 rows x 256 HD cols, f32
#pragma unroll
  for (int j = 0; j < 16; ++j)
#pragma unroll
    for (int e = 0; e < 8; ++e) acc[j][e] = 0.f;

  float mrun[8], lrun[8], sc[8];
#pragma unroll
  for (int r = 0; r < 8; ++r) { mrun[r] = -__builtin_inff(); lrun[r] = 0.f; }
  const float scale = 0.0625f; // 1/sqrt(HD)=1/16

  for (int kt = 0; kt < TT; kt += 32) {
    v8f s0, s1;
#pragma unroll
    for (int e = 0; e < 8; ++e) { s0[e] = 0.f; s1[e] = 0.f; }
    // pipelined S = Q K^T over HD
    {
      v16h aq = load_a_frag(Q, HDD, 0, 0);
      v16h b0 = load_b_frag(Kp, HDD, kt,      0);
      v16h b1 = load_b_frag(Kp, HDD, kt + 16, 0);
      for (int ks = 32; ks < HDD; ks += 32) {
        v16h aqn = load_a_frag(Q, HDD, 0, ks);
        v16h b0n = load_b_frag(Kp, HDD, kt,      ks);
        v16h b1n = load_b_frag(Kp, HDD, kt + 16, ks);
        s0 = __builtin_amdgcn_wmma_f32_16x16x32_f16(false, aq, false, b0, (short)0, s0, false, false);
        s1 = __builtin_amdgcn_wmma_f32_16x16x32_f16(false, aq, false, b1, (short)0, s1, false, false);
        aq = aqn; b0 = b0n; b1 = b1n;
      }
      s0 = __builtin_amdgcn_wmma_f32_16x16x32_f16(false, aq, false, b0, (short)0, s0, false, false);
      s1 = __builtin_amdgcn_wmma_f32_16x16x32_f16(false, aq, false, b1, (short)0, s1, false, false);
    }
    // online softmax per row (rows live in 16-lane halves per C-VGPR)
#pragma unroll
    for (int r = 0; r < 8; ++r) {
      float s0r = s0[r] * scale, s1r = s1[r] * scale;
      float mx = fmaxf(s0r, s1r);
#pragma unroll
      for (int off = 1; off <= 8; off <<= 1) mx = fmaxf(mx, __shfl_xor(mx, off));
      float mnew = fmaxf(mrun[r], mx);
      sc[r] = __expf(mrun[r] - mnew);
      float p0 = __expf(s0r - mnew);
      float p1 = __expf(s1r - mnew);
      float rs = p0 + p1;
#pragma unroll
      for (int off = 1; off <= 8; off <<= 1) rs += __shfl_xor(rs, off);
      lrun[r] = lrun[r] * sc[r] + rs;
      mrun[r] = mnew;
      int row = (l < 16) ? r : r + 8;
      int col = l & 15;
      pst[wib][row][col]      = (_Float16)p0;
      pst[wib][row][col + 16] = (_Float16)p1;
    }
#pragma unroll
    for (int j = 0; j < 16; ++j)
#pragma unroll
      for (int r = 0; r < 8; ++r) acc[j][r] *= sc[r];
    __syncthreads();
    // reload P in A-fragment layout and do P @ V
    {
      int m  = l & 15;
      int kb = (l < 16) ? 0 : 8;
      V16H pa;
      pa.h[0] = *(const v8h*)&pst[wib][m][kb];
      pa.h[1] = *(const v8h*)&pst[wib][m][kb + 16];
      v16h ap = pa.v;
#pragma unroll
      for (int nt2 = 0; nt2 < 16; ++nt2) {
        v16h bv = load_b_frag(Vt, TT, nt2 * 16, kt);  // B(k=key,n=hd) contiguous in T
        acc[nt2] = __builtin_amdgcn_wmma_f32_16x16x32_f16(false, ap, false, bv,
                                                          (short)0, acc[nt2], false, false);
      }
    }
    __syncthreads();
  }
  // epilogue: ctx[b,t, h*HD+hd] = acc / lrun
#pragma unroll
  for (int nt2 = 0; nt2 < 16; ++nt2)
#pragma unroll
    for (int r = 0; r < 8; ++r) {
      int row = (l < 16) ? r : r + 8;
      int t   = qt * 16 + row;
      int col = h * HDD + nt2 * 16 + (l & 15);
      ctx16[((size_t)b * TT + t) * DD + col] = (_Float16)(acc[nt2][r] / lrun[r]);
    }
}

// ---------------- out_proj GEMM + ReLU -> z (LDS-staged A tile) ----------------
__global__ void __launch_bounds__(128, 1)
outproj_gemm_kernel(const _Float16* __restrict__ ctx16,
                    const _Float16* __restrict__ w16,
                    const float*    __restrict__ bias,
                    _Float16* __restrict__ z16) {
  __shared__ __align__(32) _Float16 atile[16 * DD];
  int m0 = blockIdx.x * 16;
  stage_a_tile(ctx16 + (size_t)m0 * DD, atile);

  int w = threadIdx.x >> 5;
  int l = threadIdx.x & 31;
  int col0 = l & 15, rbase = (l < 16) ? 0 : 8;

  for (int g = 0; g < 2; ++g) {
    int n0 = (w * 2 + g) * 64;
    v8f acc[4];
#pragma unroll
    for (int j = 0; j < 4; ++j)
#pragma unroll
      for (int e = 0; e < 8; ++e) acc[j][e] = 0.f;
    gemm_acc<4>(atile, DD, 0, w16, DD, n0, DD, acc);
#pragma unroll
    for (int j = 0; j < 4; ++j) {
      int col = n0 + j * 16 + col0;
      float bv = bias[col];
#pragma unroll
      for (int r = 0; r < 8; ++r) {
        int row = m0 + rbase + r;
        float v = fmaxf(acc[j][r] + bv, 0.f);   // ReLU
        z16[(size_t)row * DD + col] = (_Float16)v;
      }
    }
  }
}

// ---------------- emissions GEMM (N = 32) ----------------
__global__ void __launch_bounds__(128, 1)
emis_gemm_kernel(const _Float16* __restrict__ z16,
                 const _Float16* __restrict__ wlin16,
                 const float*    __restrict__ bias,
                 float* __restrict__ emis) {
  int wave = (blockIdx.x * blockDim.x + threadIdx.x) >> 5;
  int m0 = wave * 16;
  v8f acc[2];
#pragma unroll
  for (int j = 0; j < 2; ++j)
#pragma unroll
    for (int e = 0; e < 8; ++e) acc[j][e] = 0.f;
  gemm_acc<2>(z16, DD, m0, wlin16, DD, 0, DD, acc);
  int l = threadIdx.x & 31;
  int col0 = l & 15, rbase = (l < 16) ? 0 : 8;
#pragma unroll
  for (int j = 0; j < 2; ++j) {
    int col = j * 16 + col0;
    float bv = bias[col];
#pragma unroll
    for (int r = 0; r < 8; ++r) {
      int row = m0 + rbase + r;
      emis[(size_t)row * KK + col] = acc[j][r] + bv;
    }
  }
}

// ---------------- segmentation log-softmax (2 classes) ----------------
__global__ void seg_kernel(const _Float16* __restrict__ z16,
                           const float* __restrict__ w,
                           const float* __restrict__ bias,
                           float* __restrict__ segout) {
  int row = blockIdx.x * blockDim.x + threadIdx.x;
  if (row >= BB * TT) return;
  const _Float16* zr = z16 + (size_t)row * DD;
  float s0 = bias[0], s1 = bias[1];
  for (int c = 0; c < DD; ++c) {
    float zc = (float)zr[c];
    s0 += zc * w[c];
    s1 += zc * w[DD + c];
  }
  float m = fmaxf(s0, s1);
  float ls = m + __logf(__expf(s0 - m) + __expf(s1 - m));
  segout[2 * row]     = s0 - ls;
  segout[2 * row + 1] = s1 - ls;
}

// ---------------- CRF log-likelihood: one wave per batch, lane = tag ----------------
__global__ void crf_kernel(const float* __restrict__ emis,
                           const int*   __restrict__ labels,
                           const int*   __restrict__ lengths,
                           const float* __restrict__ start,
                           const float* __restrict__ endt,
                           const float* __restrict__ trans,
                           float* __restrict__ llh,
                           float* __restrict__ msum) {
  __shared__ float tr[KK * KK];
  int b = blockIdx.x;
  int j = threadIdx.x;   // tag 0..31
  for (int i = j; i < KK * KK; i += KK) tr[i] = trans[i];
  __syncthreads();
  int len = lengths[b]; if (len < 1) len = 1;
  const float* em = emis + (size_t)b * TT * KK;
  float alpha = start[j] + em[j];
  for (int t = 1; t < TT; ++t) {
    float m = -__builtin_inff();
    for (int i = 0; i < KK; ++i) {
      float ai = __shfl(alpha, i);
      m = fmaxf(m, ai + tr[i * KK + j]);
    }
    float s = 0.f;
    for (int i = 0; i < KK; ++i) {
      float ai = __shfl(alpha, i);
      s += __expf(ai + tr[i * KK + j] - m);
    }
    float nxt = m + __logf(s) + em[(size_t)t * KK + j];
    if (t < len) alpha = nxt;
  }
  float fv = alpha + endt[j];
  float m = fv;
#pragma unroll
  for (int off = 16; off >= 1; off >>= 1) m = fmaxf(m, __shfl_xor(m, off));
  float s = __expf(fv - m);
#pragma unroll
  for (int off = 16; off >= 1; off >>= 1) s += __shfl_xor(s, off);
  float denom = m + __logf(s);
  if (j == 0) {
    const int* lab = labels + (size_t)b * TT;
    float num = start[lab[0]] + em[lab[0]];
    for (int t = 1; t < len; ++t)
      num += tr[lab[t - 1] * KK + lab[t]] + em[(size_t)t * KK + lab[t]];
    num += endt[lab[len - 1]];
    llh[b]  = num - denom;
    msum[b] = (float)len;
  }
}

__global__ void loss_kernel(const float* __restrict__ llh,
                            const float* __restrict__ msum,
                            float* __restrict__ out) {
  if (threadIdx.x == 0) {
    float a = 0.f, mm = 0.f;
    for (int t = 0; t < BB; ++t) { a += llh[t]; mm += msum[t]; }
    out[0] = -a / mm;
  }
}

// ---------------- Viterbi: forward w/ byte history, lane-0 backtrack ----------------
__global__ void viterbi_kernel(const float* __restrict__ emis,
                               const float* __restrict__ start,
                               const float* __restrict__ endt,
                               const float* __restrict__ trans,
                               unsigned char* __restrict__ hist,
                               float* __restrict__ tags) {
  __shared__ float tr[KK * KK];
  int b = blockIdx.x;
  int j = threadIdx.x;
  for (int i = j; i < KK * KK; i += KK) tr[i] = trans[i];
  __syncthreads();
  const float* em = emis + (size_t)b * TT * KK;
  unsigned char* hb = hist + (size_t)b * TT * KK;
  float score = start[j] + em[j];
  for (int t = 1; t < TT; ++t) {
    float best = -__builtin_inff(); int bi = 0;
    for (int i = 0; i < KK; ++i) {
      float v = __shfl(score, i) + tr[i * KK + j];
      if (v > best) { best = v; bi = i; }   // strict > keeps first max (argmax semantics)
    }
    hb[(size_t)t * KK + j] = (unsigned char)bi;
    score = best + em[(size_t)t * KK + j];
  }
  float bv = score + endt[j]; int bj = j;
#pragma unroll
  for (int off = 1; off < 32; off <<= 1) {
    float ov = __shfl_xor(bv, off);
    int   oj = __shfl_xor(bj, off);
    if (ov > bv || (ov == bv && oj < bj)) { bv = ov; bj = oj; }
  }
  if (j == 0) {
    int tag = bj;
    float* tb = tags + (size_t)b * TT;
    tb[TT - 1] = (float)tag;
    for (int t = TT - 1; t >= 1; --t) {
      tag = hb[(size_t)t * KK + tag];
      tb[t - 1] = (float)tag;
    }
  }
}

// ---------------- launcher ----------------
extern "C" void kernel_launch(void* const* d_in, const int* in_sizes, int n_in,
                              void* d_out, int out_size, void* d_ws, size_t ws_size,
                              hipStream_t stream) {
  const float* x       = (const float*)d_in[0];
  const int*   labels  = (const int*)d_in[2];
  const int*   lengths = (const int*)d_in[3];
  const float* in_w    = (const float*)d_in[4];
  const float* in_b    = (const float*)d_in[5];
  const float* out_w   = (const float*)d_in[6];
  const float* out_b   = (const float*)d_in[7];
  const float* lin_w   = (const float*)d_in[8];
  const float* lin_b   = (const float*)d_in[9];
  const float* seg_w   = (const float*)d_in[10];
  const float* seg_b   = (const float*)d_in[11];
  const float* start_t = (const float*)d_in[12];
  const float* end_t   = (const float*)d_in[13];
  const float* trans   = (const float*)d_in[14];

  char* ws = (char*)d_ws;
  size_t off = 0;
  auto alloc = [&](size_t bytes) -> void* {
    void* p = (void*)(ws + off);
    off += (bytes + 255) & ~(size_t)255;
    return p;
  };
  _Float16* x16    = (_Float16*)alloc((size_t)BB * TT * DD * 2);
  _Float16* win16  = (_Float16*)alloc((size_t)3 * DD * DD * 2);
  _Float16* wout16 = (_Float16*)alloc((size_t)DD * DD * 2);
  _Float16* wlin16 = (_Float16*)alloc((size_t)KK * DD * 2);
  _Float16* q16    = (_Float16*)alloc((size_t)BB * TT * DD * 2);
  _Float16* k16    = (_Float16*)alloc((size_t)BB * TT * DD * 2);
  _Float16* vt16   = (_Float16*)alloc((size_t)BB * TT * DD * 2);
  _Float16* ctx16  = (_Float16*)alloc((size_t)BB * TT * DD * 2);
  _Float16* z16    = (_Float16*)alloc((size_t)BB * TT * DD * 2);
  float*    emis   = (float*)alloc((size_t)BB * TT * KK * 4);
  unsigned char* hist = (unsigned char*)alloc((size_t)BB * TT * KK);
  float*    llh    = (float*)alloc(BB * 4);
  float*    msum   = (float*)alloc(BB * 4);

  float* tags_out = (float*)d_out;
  float* seg_out  = (float*)d_out + (size_t)BB * TT;
  float* loss_out = (float*)d_out + (size_t)BB * TT + (size_t)BB * TT * 2;

  int n;
  n = BB * TT * DD; cvt_f32_f16_kernel<<<(n + 255) / 256, 256, 0, stream>>>(x, x16, n);
  n = 3 * DD * DD;  cvt_f32_f16_kernel<<<(n + 255) / 256, 256, 0, stream>>>(in_w, win16, n);
  n = DD * DD;      cvt_f32_f16_kernel<<<(n + 255) / 256, 256, 0, stream>>>(out_w, wout16, n);
  n = KK * DD;      cvt_f32_f16_kernel<<<(n + 255) / 256, 256, 0, stream>>>(lin_w, wlin16, n);

  // QKV: one block per 16-row m-tile (A staged in LDS via async DMA)
  qkv_gemm_kernel<<<2048, 128, 0, stream>>>(x16, win16, in_b, q16, k16, vt16);
  // attention: B*H*(T/16) = 4096 waves
  attn_kernel<<<1024, 128, 0, stream>>>(q16, k16, vt16, ctx16);
  // out_proj: one block per m-tile
  outproj_gemm_kernel<<<2048, 128, 0, stream>>>(ctx16, wout16, out_b, z16);
  // emissions: 2048 waves
  emis_gemm_kernel<<<512, 128, 0, stream>>>(z16, wlin16, lin_b, emis);
  // seg log-softmax
  seg_kernel<<<128, 256, 0, stream>>>(z16, seg_w, seg_b, seg_out);
  // CRF + loss
  crf_kernel<<<BB, 32, 0, stream>>>(emis, labels, lengths, start_t, end_t, trans, llh, msum);
  loss_kernel<<<1, 32, 0, stream>>>(llh, msum, loss_out);
  // Viterbi decode
  viterbi_kernel<<<BB, 32, 0, stream>>>(emis, start_t, end_t, trans, hist, tags_out);
}
